// LocalAttention_2886218023040
// MI455X (gfx1250) — compile-verified, matrix operand order checked by reading, not run on hardware
//
#include <hip/hip_runtime.h>
#include <hip/hip_bf16.h>

#define H_   96
#define W_   96
#define HW   9216
#define P_   225
#define WIN  15
#define RAD  7
#define DATT 64
#define DVU  1024
#define VPW  112      // padded x-width of vpad (8 + 96 + 8)

typedef __bf16 bf16;
typedef __attribute__((ext_vector_type(16))) __bf16          bf16x16;
typedef __attribute__((ext_vector_type(8)))  float           floatx8;
typedef __attribute__((ext_vector_type(8)))  unsigned short  ushort8;

// Fragment = v16bf whose two ushort8 halves are the two contiguous K-runs
// (ISA 7.12.2: element i -> K = (i&7) + 8g + 16*(i>>3), g = lane>>4).
union FragBF { bf16x16 v; ushort8 h[2]; };
union Pack16 { ushort8 h[2]; unsigned short e[16]; };

__device__ __forceinline__ bf16 f2bf(float f) {
  unsigned u = __builtin_bit_cast(unsigned, f);
  unsigned r = u + 0x7FFFu + ((u >> 16) & 1u);
  return __builtin_bit_cast(bf16, (unsigned short)(r >> 16));
}
__device__ __forceinline__ float bf2f(bf16 b) {
  unsigned u = ((unsigned)__builtin_bit_cast(unsigned short, b)) << 16;
  return __builtin_bit_cast(float, u);
}

// ---------------- f32 -> bf16 convert (grid-stride) ----------------
__global__ void cvt_f32_bf16(const float* __restrict__ src, bf16* __restrict__ dst, int n) {
  for (int i = blockIdx.x * blockDim.x + threadIdx.x; i < n; i += gridDim.x * blockDim.x)
    dst[i] = f2bf(src[i]);
}

// v (1024,96,96) f32 -> zero-padded bf16 (1024,96,112), x shifted by +8
__global__ void cvt_v_pad(const float* __restrict__ v, bf16* __restrict__ vpad) {
  const int n = DVU * H_ * VPW;
  for (int i = blockIdx.x * blockDim.x + threadIdx.x; i < n; i += gridDim.x * blockDim.x) {
    const int px = i % VPW;
    const int cy = i / VPW;                  // c*96 + y
    const int x = px - 8;
    bf16 val = __builtin_bit_cast(bf16, (unsigned short)0);
    if (x >= 0 && x < W_) val = f2bf(v[(size_t)cy * W_ + x]);
    vpad[i] = val;
  }
}

// w_dw (1024, 25) -> (25, 1024) for coalesced per-tap weight loads
__global__ void transpose_wdw(const float* __restrict__ w_dw, float* __restrict__ wdwT) {
  const int i = blockIdx.x * 256 + threadIdx.x;
  if (i < DVU * 25) {
    const int c = i / 25, t = i % 25;
    wdwT[t * DVU + c] = w_dw[i];
  }
}

// ---------------- corr + rel -> scores (f32 VALU, ~0.5 GF) ----------------
__global__ void scores_kernel(const float* __restrict__ q, const float* __restrict__ kg,
                              const float* __restrict__ w_rel, const float* __restrict__ b_rel,
                              float* __restrict__ scores) {
  __shared__ float qs[DATT][128];            // 32 KB
  const int t = threadIdx.x;
  const int pbase = blockIdx.x * 128;
  for (int c = 0; c < DATT; ++c) qs[c][t] = q[c * HW + pbase + t];
  __syncthreads();
  const int p = pbase + t;
  const int y = p / W_, x = p % W_;
  for (int o = 0; o < P_; ++o) {
    const int dy = o / WIN - RAD, dx = o % WIN - RAD;
    const int yn = y + dy, xn = x + dx;
    float sr = 0.f;
    const float* wr = w_rel + o * DATT;      // uniform -> scalar loads
    #pragma unroll 8
    for (int c = 0; c < DATT; ++c) sr += wr[c] * qs[c][t];
    float sc = 0.f;
    if (yn >= 0 && yn < H_ && xn >= 0 && xn < W_) {
      const float* kp = kg + yn * W_ + xn;
      #pragma unroll 8
      for (int c = 0; c < DATT; ++c) sc += qs[c][t] * kp[c * HW];
    }
    scores[o * HW + p] = sc * 0.125f + sr + b_rel[o];
  }
}

// ---------------- softmax; emit PRE-MASKED attn in (p, dy, 16) layout ----------------
// attn_w[(p*15 + dy)*16 + dx] = softmax weight, 0 for OOB neighbors; slot 15 = 0.
// Each 16-slot run is 32B aligned -> band staging is 2x b128, branch-free.
__global__ void softmax_kernel(float* __restrict__ scores, bf16* __restrict__ attn_w) {
  const int p = blockIdx.x * 256 + threadIdx.x;
  const int y = p / W_, x = p % W_;
  float m = -1e30f;
  for (int o = 0; o < P_; ++o) m = fmaxf(m, scores[o * HW + p]);
  float s = 0.f;
  for (int o = 0; o < P_; ++o) {
    float e = __expf(scores[o * HW + p] - m);
    scores[o * HW + p] = e;
    s += e;
  }
  const float r = 1.f / s;
  for (int dy = 0; dy < WIN; ++dy) {
    const int yn = y + dy - RAD;
    const bool rok = (yn >= 0) && (yn < H_);
    Pack16 u;
    #pragma unroll
    for (int dx = 0; dx < WIN; ++dx) {
      const int xn = x + dx - RAD;
      const float val = (rok && xn >= 0 && xn < W_)
                          ? scores[(dy * WIN + dx) * HW + p] * r : 0.f;
      u.e[dx] = __builtin_bit_cast(unsigned short, f2bf(val));
    }
    u.e[15] = 0;
    ushort8* dst = (ushort8*)(attn_w + ((size_t)p * WIN + dy) * 16);
    dst[0] = u.h[0];
    dst[1] = u.h[1];
  }
}

// ---------------- aggregation: banded WMMA ----------------
// Band columns kk=0..31 <-> absolute xa = x0-8+kk (padded coord px = x0+kk).
// A(16x32): A[m][m+1+d] = attn_w[(y*96+x0+m, dy)][d]; B(32x16) = vpad columns.
__global__ void agg_kernel(const bf16* __restrict__ attn_w, const bf16* __restrict__ vpad,
                           bf16* __restrict__ aggb_p) {
  __shared__ __align__(16) bf16 Atile[16][32];       // 1 KB, shared by all 8 waves
  __shared__ __align__(16) bf16 Tt[8][16][16];       // 4 KB wave-local transpose
  const int tid  = threadIdx.x;
  const int lane = tid & 31, w = tid >> 5;
  const int x0 = blockIdx.x * 16, y = blockIdx.y;
  const int g = lane >> 4, mn = lane & 15;
  floatx8 acc[8] = {};
  for (int dy = 0; dy < WIN; ++dy) {
    const int row = y + dy - RAD;                    // uniform per block
    const bool rowok = (row >= 0) && (row < H_);
    __syncthreads();
    if (rowok && tid < 16) {                         // stage band row m = tid, branch-free
      const int m = tid;
      const ushort8 z = {0, 0, 0, 0, 0, 0, 0, 0};
      *(ushort8*)&Atile[m][0]  = z;  *(ushort8*)&Atile[m][8]  = z;
      *(ushort8*)&Atile[m][16] = z;  *(ushort8*)&Atile[m][24] = z;
      Pack16 u;                                      // 2x global_load_b128 (pre-masked)
      const ushort8* ap =
          (const ushort8*)(attn_w + ((size_t)(y * W_ + x0 + m) * WIN + dy) * 16);
      u.h[0] = ap[0];
      u.h[1] = ap[1];
      #pragma unroll
      for (int d = 0; d < 16; ++d)                   // kk = m+1 .. m+16 (<= 31)
        Atile[m][m + 1 + d] = __builtin_bit_cast(bf16, u.e[d]);
    }
    __syncthreads();
    if (!rowok) continue;
    FragBF fa;                                       // 2x ds_load_b128
    fa.h[0] = *(const ushort8*)&Atile[mn][g << 3];
    fa.h[1] = *(const ushort8*)&Atile[mn][16 + (g << 3)];
    #pragma unroll
    for (int j = 0; j < 8; ++j) {
      const int c0 = ((w << 3) + j) << 4;            // wave covers channels [w*128, +128)
      const bf16* vrow = vpad + ((size_t)(c0 + mn) * H_ + row) * VPW + x0;
      FragBF fb;                                     // 2x global_load_b128 (padded)
      fb.h[0] = *(const ushort8*)(vrow + (g << 3));
      fb.h[1] = *(const ushort8*)(vrow + 16 + (g << 3));
      acc[j] = __builtin_amdgcn_wmma_f32_16x16x32_bf16(
          false, fa.v, false, fb.v, (short)0, acc[j], false, false);
    }
  }
  // wave-local LDS transpose -> coalesced 2x16B global stores per lane
  #pragma unroll
  for (int j = 0; j < 8; ++j) {
    const int c0 = ((w << 3) + j) << 4;
    #pragma unroll
    for (int r = 0; r < 8; ++r) Tt[w][r + (g << 3)][mn] = f2bf(acc[j][r]);
    if (lane < 16) {                                 // in-order DS ops within wave
      const int m = lane;
      bf16* dst = aggb_p + (size_t)(y * W_ + x0 + m) * DVU + c0;
      *(ushort8*)dst       = *(const ushort8*)&Tt[w][m][0];
      *(ushort8*)(dst + 8) = *(const ushort8*)&Tt[w][m][8];
    }
  }
}

// ---------------- depthwise 5x5 conv, position-major (uniform branches) ----------------
__global__ void dwconv_kernel(const bf16* __restrict__ aggb_p, const float* __restrict__ wdwT,
                              bf16* __restrict__ yb_p) {
  const int p = blockIdx.x;                          // one position per block
  const int y = p / W_, x = p % W_;
  const int t = threadIdx.x;
  for (int chunk = 0; chunk < 4; ++chunk) {
    const int c = chunk * 256 + t;
    float wreg[25];
    #pragma unroll
    for (int i = 0; i < 25; ++i) wreg[i] = wdwT[i * DVU + c];  // coalesced per tap
    float s = 0.f;
    #pragma unroll
    for (int ky = 0; ky < 5; ++ky) {
      const int yy = y + ky - 2;
      if (yy < 0 || yy >= H_) continue;              // uniform
      #pragma unroll
      for (int kx = 0; kx < 5; ++kx) {
        const int xx = x + kx - 2;
        if (xx < 0 || xx >= W_) continue;            // uniform
        s += wreg[ky * 5 + kx] * bf2f(aggb_p[(size_t)(yy * W_ + xx) * DVU + c]);
      }
    }
    yb_p[(size_t)p * DVU + c] = f2bf(s);
  }
}

// ---------------- projection GEMM: (9216x1024) @ (1024x1024)^T, BK=64 ----------------
__global__ void proj_kernel(const bf16* __restrict__ yb_p, const bf16* __restrict__ wpb,
                            const float* __restrict__ b_proj, float* __restrict__ out) {
  __shared__ __align__(16) bf16 Asm[64][64];         // 8 KB, [m][k] (K-contiguous)
  __shared__ __align__(16) bf16 Bs[128][64];         // 16 KB, [n][k]
  const int tid  = threadIdx.x;
  const int lane = tid & 31, w = tid >> 5;
  const int wm = w >> 2, wn = w & 3;                 // 2x4 wave grid
  const int g = lane >> 4, ln = lane & 15;
  const int pm0 = blockIdx.x * 64;
  const int n0  = blockIdx.y * 128;
  floatx8 acc[2][2] = {};
  for (int k0 = 0; k0 < DVU; k0 += 64) {
    __syncthreads();
    { // A: 64m x 64k, 16 contiguous k per thread (2x b128 each way)
      const int m = tid >> 2, kc = (tid & 3) << 4;
      const ushort8* src = (const ushort8*)(yb_p + (size_t)(pm0 + m) * DVU + k0 + kc);
      ushort8* dst = (ushort8*)&Asm[m][kc];
      dst[0] = src[0]; dst[1] = src[1];
    }
    { // B: 128n x 64k, 32 contiguous k per thread (4x b128)
      const int n = tid >> 1, kc = (tid & 1) << 5;
      const ushort8* src = (const ushort8*)(wpb + (size_t)(n0 + n) * DVU + k0 + kc);
      ushort8* dst = (ushort8*)&Bs[n][kc];
      dst[0] = src[0]; dst[1] = src[1]; dst[2] = src[2]; dst[3] = src[3];
    }
    __syncthreads();
    if (k0 + 64 < DVU) {                             // global_prefetch_b8 next K-slab
      __builtin_prefetch(yb_p + (size_t)(pm0 + (tid >> 2)) * DVU + k0 + 64, 0, 1);
      __builtin_prefetch(wpb + (size_t)(n0 + (tid >> 1)) * DVU + k0 + 64, 0, 1);
    }
    #pragma unroll
    for (int kk = 0; kk < 64; kk += 32) {
      FragBF a[2], b[2];
      #pragma unroll
      for (int mf = 0; mf < 2; ++mf) {
        const bf16* ar = &Asm[wm * 32 + mf * 16 + ln][kk];
        a[mf].h[0] = *(const ushort8*)(ar + (g << 3));
        a[mf].h[1] = *(const ushort8*)(ar + 16 + (g << 3));
      }
      #pragma unroll
      for (int nf = 0; nf < 2; ++nf) {
        const bf16* br = &Bs[wn * 32 + nf * 16 + ln][kk];
        b[nf].h[0] = *(const ushort8*)(br + (g << 3));
        b[nf].h[1] = *(const ushort8*)(br + 16 + (g << 3));
      }
      #pragma unroll
      for (int mf = 0; mf < 2; ++mf)
        #pragma unroll
        for (int nf = 0; nf < 2; ++nf)
          acc[mf][nf] = __builtin_amdgcn_wmma_f32_16x16x32_bf16(
              false, a[mf].v, false, b[nf].v, (short)0, acc[mf][nf], false, false);
    }
  }
  #pragma unroll
  for (int mf = 0; mf < 2; ++mf) {
    #pragma unroll
    for (int nf = 0; nf < 2; ++nf) {
      const int cout = n0 + wn * 32 + nf * 16 + ln;
      const float bp = b_proj[cout];
      const int p0 = pm0 + wm * 32 + mf * 16 + (g << 3);   // lane's 8 consecutive p
      float4* dst = (float4*)(out + (size_t)cout * HW + p0);
      float4 o0, o1;
      o0.x = acc[mf][nf][0] + bp; o0.y = acc[mf][nf][1] + bp;
      o0.z = acc[mf][nf][2] + bp; o0.w = acc[mf][nf][3] + bp;
      o1.x = acc[mf][nf][4] + bp; o1.y = acc[mf][nf][5] + bp;
      o1.z = acc[mf][nf][6] + bp; o1.w = acc[mf][nf][7] + bp;
      dst[0] = o0; dst[1] = o1;
    }
  }
}

extern "C" void kernel_launch(void* const* d_in, const int* in_sizes, int n_in,
                              void* d_out, int out_size, void* d_ws, size_t ws_size,
                              hipStream_t stream) {
  (void)in_sizes; (void)n_in; (void)out_size; (void)ws_size;
  const float* q      = (const float*)d_in[0];
  const float* k      = (const float*)d_in[1];
  const float* v      = (const float*)d_in[2];
  const float* w_rel  = (const float*)d_in[3];
  const float* b_rel  = (const float*)d_in[4];
  const float* w_dw   = (const float*)d_in[5];
  const float* w_proj = (const float*)d_in[6];
  const float* b_proj = (const float*)d_in[7];
  float* out = (float*)d_out;

  char* ws = (char*)d_ws;
  size_t off = 0;
  auto carve = [&](size_t bytes) {
    void* p = (void*)(ws + off);
    off += (bytes + 255) & ~(size_t)255;
    return p;
  };
  float* scores = (float*)carve((size_t)P_ * HW * 4);        //  8.3 MB
  bf16*  attn_w = (bf16*) carve((size_t)HW * WIN * 16 * 2);  //  4.4 MB
  bf16*  vpad   = (bf16*) carve((size_t)DVU * H_ * VPW * 2); // 22.0 MB
  bf16*  aggb_p = (bf16*) carve((size_t)HW * DVU * 2);       // 18.9 MB
  bf16*  yb_p   = (bf16*) carve((size_t)HW * DVU * 2);       // 18.9 MB
  bf16*  wpb    = (bf16*) carve((size_t)DVU * DVU * 2);      //  2.1 MB
  float* wdwT   = (float*)carve((size_t)25 * DVU * 4);       //  0.1 MB

  cvt_v_pad<<<1024, 256, 0, stream>>>(v, vpad);
  cvt_f32_bf16<<<512, 256, 0, stream>>>(w_proj, wpb, DVU * DVU);
  transpose_wdw<<<(DVU * 25 + 255) / 256, 256, 0, stream>>>(w_dw, wdwT);
  scores_kernel<<<HW / 128, 128, 0, stream>>>(q, k, w_rel, b_rel, scores);
  softmax_kernel<<<HW / 256, 256, 0, stream>>>(scores, attn_w);
  agg_kernel<<<dim3(W_ / 16, H_), 256, 0, stream>>>(attn_w, vpad, aggb_p);
  dwconv_kernel<<<HW, 256, 0, stream>>>(aggb_p, wdwT, yb_p);
  proj_kernel<<<dim3(HW / 64, DVU / 128), 256, 0, stream>>>(yb_p, wpb, b_proj, out);
}
